// MoEBlock_62732292325764
// MI455X (gfx1250) — compile-verified, matrix operand order checked by reading, not run on hardware
//
#include <hip/hip_runtime.h>
#include <hip/hip_bf16.h>
#include <cmath>
#include <stdint.h>

typedef _Float16 half_t;
typedef __attribute__((ext_vector_type(16))) _Float16 v16h;
typedef __attribute__((ext_vector_type(8)))  float    v8f;

#define N_TOK (16 * 1024)
#define HID   768
#define LDSTR 40   // LDS row stride in halves (32 + 8 pad)

// ---------------- fp32 -> fp16 convert ----------------
__global__ __launch_bounds__(256) void cvt_f32_f16(const float* __restrict__ src,
                                                   half_t* __restrict__ dst, int n) {
  int stride = gridDim.x * blockDim.x;
  for (int i = blockIdx.x * blockDim.x + threadIdx.x; i < n; i += stride)
    dst[i] = (half_t)src[i];
}

// ---------------- router: logits, softmax, top-2 renorm, usage-loss partials ----------------
__global__ __launch_bounds__(256) void router_kernel(const float* __restrict__ x,
                                                     const float* __restrict__ pad,
                                                     const float* __restrict__ gate,
                                                     float* __restrict__ logits,
                                                     float* __restrict__ dw,
                                                     float* __restrict__ part) {
  __shared__ float sC[8], sV[8];
  int wid = threadIdx.x >> 5, lane = threadIdx.x & 31;
  int tok = blockIdx.x * 8 + wid;
  const float* xr = x + (size_t)tok * HID;
  float l0 = 0.f, l1 = 0.f, l2 = 0.f;
  for (int k = lane; k < HID; k += 32) {
    float xv = xr[k];
    l0 += xv * gate[0 * HID + k];
    l1 += xv * gate[1 * HID + k];
    l2 += xv * gate[2 * HID + k];
  }
#pragma unroll
  for (int o = 16; o > 0; o >>= 1) {
    l0 += __shfl_xor(l0, o, 32);
    l1 += __shfl_xor(l1, o, 32);
    l2 += __shfl_xor(l2, o, 32);
  }
  if (lane == 0) {
    logits[tok * 3 + 0] = l0;
    logits[tok * 3 + 1] = l1;
    logits[tok * 3 + 2] = l2;
    float m = fmaxf(l0, fmaxf(l1, l2));
    float e0 = __expf(l0 - m), e1 = __expf(l1 - m), e2 = __expf(l2 - m);
    float inv = 1.f / (e0 + e1 + e2);
    float p0 = e0 * inv, p1 = e1 * inv, p2 = e2 * inv;
    float valid = (pad[tok] != -INFINITY) ? 1.f : 0.f;
    // costs = (dims/sum)^2 = (0, 1/9, 4/9)
    const float c1 = 1.f / 9.f, c2 = 4.f / 9.f;
    sC[wid] = (p1 * c1 + p2 * c2) * valid;
    sV[wid] = valid;
    // drop argmin (later index wins ties, matching top_k keeping earlier indices)
    float pm = p2; int amin = 2;
    if (p1 < pm) { pm = p1; amin = 1; }
    if (p0 < pm) { pm = p0; amin = 0; }
    float rs = 1.f / (1.f - pm);
    dw[tok * 3 + 0] = (amin == 0) ? 0.f : p0 * rs;
    dw[tok * 3 + 1] = (amin == 1) ? 0.f : p1 * rs;
    dw[tok * 3 + 2] = (amin == 2) ? 0.f : p2 * rs;
  }
  __syncthreads();
  if (threadIdx.x == 0) {
    float c = 0.f, v = 0.f;
    for (int i = 0; i < 8; ++i) { c += sC[i]; v += sV[i]; }
    part[blockIdx.x * 2 + 0] = c;
    part[blockIdx.x * 2 + 1] = v;
  }
}

__global__ __launch_bounds__(256) void loss_reduce(const float* __restrict__ part, int nb,
                                                   float* __restrict__ out_loss) {
  __shared__ float sc[256], sv[256];
  float c = 0.f, v = 0.f;
  for (int i = threadIdx.x; i < nb; i += 256) { c += part[2 * i]; v += part[2 * i + 1]; }
  sc[threadIdx.x] = c; sv[threadIdx.x] = v;
  __syncthreads();
  for (int o = 128; o > 0; o >>= 1) {
    if (threadIdx.x < o) { sc[threadIdx.x] += sc[threadIdx.x + o]; sv[threadIdx.x] += sv[threadIdx.x + o]; }
    __syncthreads();
  }
  if (threadIdx.x == 0) out_loss[0] = sc[0] / sv[0];
}

// ---------------- out = dw[:,0] * x  (expert 0 is identity) ----------------
__global__ __launch_bounds__(256) void init_out(const float* __restrict__ x,
                                                const float* __restrict__ dw,
                                                float* __restrict__ out, int n) {
  int stride = gridDim.x * blockDim.x;
  for (int i = blockIdx.x * blockDim.x + threadIdx.x; i < n; i += stride) {
    int tok = i / HID;
    out[i] = dw[tok * 3] * x[i];
  }
}

// ---------------- WMMA fragment load from LDS ----------------
// 16-bit 16x32 operand, ISA layout: lanes 0-15 -> M/N=lane, K in {kb..kb+7, kb+16..kb+23}, kb = (lane>=16)?8:0
__device__ __forceinline__ v16h ld_frag(const half_t* base, int rowBase) {
  int lane = threadIdx.x & 31;
  int row = rowBase + (lane & 15);
  int kb2 = (lane >> 4) << 2;  // kb in dwords (0 or 4)
  const unsigned int* p = (const unsigned int*)(base + row * LDSTR);
  union { v16h h; unsigned int u[8]; } f;
#pragma unroll
  for (int i = 0; i < 4; ++i) {
    f.u[i]     = p[kb2 + i];      // K = kb + 2i, kb + 2i + 1
    f.u[4 + i] = p[8 + kb2 + i];  // K = 16 + kb + 2i, ...
  }
  return f.h;
}

// fast silu(a)*b using hardware reciprocal (v_rcp_f32) instead of IEEE divide
__device__ __forceinline__ float swiglu_act(float a, float b) {
  float sig = __builtin_amdgcn_rcpf(1.f + __expf(-a));
  return a * sig * b;
}

// ---------------- GEMM #1 with fused SwiGLU: H = silu(x Win1^T) * (x Win2^T) ----------------
// A: [N_TOK, 768] f16, Win: [2*hdim, 768] f16, Hout: [N_TOK, hdim] f16
// Double-buffered LDS, register prefetch two tiles ahead, one barrier per K-step.
__global__ __launch_bounds__(256) void gemm_swiglu(const half_t* __restrict__ A,
                                                   const half_t* __restrict__ Win,
                                                   half_t* __restrict__ Hout, int hdim) {
  __shared__ half_t sA[2][128 * LDSTR];
  __shared__ half_t sB[2][128 * LDSTR];
  const int K = HID;
  const int KSTEPS = K / 32;
  int mb = blockIdx.y * 128;
  int nb = blockIdx.x * 64;
  int tid = threadIdx.x;
  int wid = tid >> 5, lane = tid & 31;
  int wm = (wid >> 1) * 32;   // wave M offset within 128
  int wn = (wid & 1) * 32;    // wave N offset within 64
  v8f zero = {0.f, 0.f, 0.f, 0.f, 0.f, 0.f, 0.f, 0.f};
  v8f acc1[2][2], acc2[2][2];
#pragma unroll
  for (int i = 0; i < 2; ++i)
#pragma unroll
    for (int j = 0; j < 2; ++j) { acc1[i][j] = zero; acc2[i][j] = zero; }

  int lr = tid >> 1;           // 0..127 (tile row)
  int lc = (tid & 1) * 16;     // half-chunk
  const half_t* gA = A + (size_t)(mb + lr) * K + lc;
  int brow = (lr < 64) ? (nb + lr) : (hdim + nb + (lr - 64));
  const half_t* gB = Win + (size_t)brow * K + lc;

  // prologue: load tile 0, store into buf 0, prefetch tile 1 into regs
  uint4 ra0, ra1, rb0, rb1;
  {
    const uint4* pa = (const uint4*)gA;
    const uint4* pb = (const uint4*)gB;
    ra0 = pa[0]; ra1 = pa[1]; rb0 = pb[0]; rb1 = pb[1];
    uint4* qa = (uint4*)(&sA[0][lr * LDSTR + lc]);
    uint4* qb = (uint4*)(&sB[0][lr * LDSTR + lc]);
    qa[0] = ra0; qa[1] = ra1; qb[0] = rb0; qb[1] = rb1;
  }
  if (KSTEPS > 1) {
    const uint4* pa = (const uint4*)(gA + 32);
    const uint4* pb = (const uint4*)(gB + 32);
    ra0 = pa[0]; ra1 = pa[1]; rb0 = pb[0]; rb1 = pb[1];
  }

  for (int ks = 0; ks < KSTEPS; ++ks) {
    __syncthreads();
    int nxt = (ks + 1) & 1;
    if (ks + 1 < KSTEPS) {
      uint4* qa = (uint4*)(&sA[nxt][lr * LDSTR + lc]);
      uint4* qb = (uint4*)(&sB[nxt][lr * LDSTR + lc]);
      qa[0] = ra0; qa[1] = ra1; qb[0] = rb0; qb[1] = rb1;
    }
    if (ks + 2 < KSTEPS) {
      const uint4* pa = (const uint4*)(gA + (ks + 2) * 32);
      const uint4* pb = (const uint4*)(gB + (ks + 2) * 32);
      ra0 = pa[0]; ra1 = pa[1]; rb0 = pb[0]; rb1 = pb[1];
    }
    const half_t* cA = sA[ks & 1];
    const half_t* cB = sB[ks & 1];
    v16h af[2], b1f[2], b2f[2];
    af[0]  = ld_frag(cA, wm);
    af[1]  = ld_frag(cA, wm + 16);
    b1f[0] = ld_frag(cB, wn);
    b1f[1] = ld_frag(cB, wn + 16);
    b2f[0] = ld_frag(cB, 64 + wn);
    b2f[1] = ld_frag(cB, 64 + wn + 16);
#pragma unroll
    for (int i = 0; i < 2; ++i)
#pragma unroll
      for (int j = 0; j < 2; ++j) {
        acc1[i][j] = __builtin_amdgcn_wmma_f32_16x16x32_f16(
            false, af[i], false, b1f[j], (short)0, acc1[i][j], false, false);
        acc2[i][j] = __builtin_amdgcn_wmma_f32_16x16x32_f16(
            false, af[i], false, b2f[j], (short)0, acc2[i][j], false, false);
      }
  }
  // SwiGLU epilogue -> f16 H
#pragma unroll
  for (int i = 0; i < 2; ++i)
#pragma unroll
    for (int j = 0; j < 2; ++j) {
      int c  = nb + wn + j * 16 + (lane & 15);
      int r0 = mb + wm + i * 16 + ((lane >> 4) << 3);
#pragma unroll
      for (int r = 0; r < 8; ++r) {
        float hv = swiglu_act(acc1[i][j][r], acc2[i][j][r]);
        Hout[(size_t)(r0 + r) * hdim + c] = (half_t)hv;
      }
    }
}

// ---------------- GEMM #2 + weighted combine: out += dw[:,e] * (H Wout^T) ----------------
// A: [N_TOK, hdim] f16, Wout: [768, hdim] f16
__global__ __launch_bounds__(256) void gemm_combine(const half_t* __restrict__ A,
                                                    const half_t* __restrict__ Wout,
                                                    const float* __restrict__ dw,
                                                    float* __restrict__ out,
                                                    int hdim, int expert) {
  __shared__ half_t sA[2][128 * LDSTR];
  __shared__ half_t sB[2][64 * LDSTR];
  const int KSTEPS = hdim / 32;
  int mb = blockIdx.y * 128;
  int nb = blockIdx.x * 64;
  int tid = threadIdx.x;
  int wid = tid >> 5, lane = tid & 31;
  int wm = (wid >> 1) * 32;
  int wn = (wid & 1) * 32;
  v8f zero = {0.f, 0.f, 0.f, 0.f, 0.f, 0.f, 0.f, 0.f};
  v8f acc[2][2];
#pragma unroll
  for (int i = 0; i < 2; ++i)
#pragma unroll
    for (int j = 0; j < 2; ++j) acc[i][j] = zero;

  int lr = tid >> 1;           // A tile row 0..127
  int lc = (tid & 1) * 16;
  const half_t* gA = A + (size_t)(mb + lr) * hdim + lc;
  int rb = tid >> 2;           // B tile row 0..63
  int cb = (tid & 3) * 8;
  const half_t* gB = Wout + (size_t)(nb + rb) * hdim + cb;

  // prologue
  uint4 ra0, ra1, rbv;
  {
    const uint4* pa = (const uint4*)gA;
    ra0 = pa[0]; ra1 = pa[1];
    rbv = *(const uint4*)gB;
    uint4* qa = (uint4*)(&sA[0][lr * LDSTR + lc]);
    qa[0] = ra0; qa[1] = ra1;
    *(uint4*)(&sB[0][rb * LDSTR + cb]) = rbv;
  }
  if (KSTEPS > 1) {
    const uint4* pa = (const uint4*)(gA + 32);
    ra0 = pa[0]; ra1 = pa[1];
    rbv = *(const uint4*)(gB + 32);
  }

  for (int ks = 0; ks < KSTEPS; ++ks) {
    __syncthreads();
    int nxt = (ks + 1) & 1;
    if (ks + 1 < KSTEPS) {
      uint4* qa = (uint4*)(&sA[nxt][lr * LDSTR + lc]);
      qa[0] = ra0; qa[1] = ra1;
      *(uint4*)(&sB[nxt][rb * LDSTR + cb]) = rbv;
    }
    if (ks + 2 < KSTEPS) {
      const uint4* pa = (const uint4*)(gA + (ks + 2) * 32);
      ra0 = pa[0]; ra1 = pa[1];
      rbv = *(const uint4*)(gB + (ks + 2) * 32);
    }
    const half_t* cA = sA[ks & 1];
    const half_t* cB = sB[ks & 1];
    v16h af[2], bf[2];
    af[0] = ld_frag(cA, wm);
    af[1] = ld_frag(cA, wm + 16);
    bf[0] = ld_frag(cB, wn);
    bf[1] = ld_frag(cB, wn + 16);
#pragma unroll
    for (int i = 0; i < 2; ++i)
#pragma unroll
      for (int j = 0; j < 2; ++j)
        acc[i][j] = __builtin_amdgcn_wmma_f32_16x16x32_f16(
            false, af[i], false, bf[j], (short)0, acc[i][j], false, false);
  }
#pragma unroll
  for (int i = 0; i < 2; ++i)
#pragma unroll
    for (int j = 0; j < 2; ++j) {
      int c  = nb + wn + j * 16 + (lane & 15);
      int r0 = mb + wm + i * 16 + ((lane >> 4) << 3);
#pragma unroll
      for (int r = 0; r < 8; ++r) {
        int row = r0 + r;
        float w = dw[row * 3 + expert];
        size_t idx = (size_t)row * HID + c;
        out[idx] += w * acc[i][j][r];
      }
    }
}

// ---------------- host launch ----------------
extern "C" void kernel_launch(void* const* d_in, const int* in_sizes, int n_in,
                              void* d_out, int out_size, void* d_ws, size_t ws_size,
                              hipStream_t stream) {
  (void)in_sizes; (void)n_in; (void)out_size; (void)ws_size;
  const float* x    = (const float*)d_in[0];
  const float* pad  = (const float*)d_in[2];
  const float* gate = (const float*)d_in[3];
  const float* w1i  = (const float*)d_in[4];  // [2048, 768]
  const float* w1o  = (const float*)d_in[5];  // [768, 1024]
  const float* w2i  = (const float*)d_in[6];  // [4096, 768]
  const float* w2o  = (const float*)d_in[7];  // [768, 2048]

  float* out    = (float*)d_out;                      // [N_TOK, 768]
  float* logits = out + (size_t)N_TOK * HID;          // [N_TOK, 3]
  float* loss   = logits + (size_t)N_TOK * 3;         // scalar

  // workspace carve-up
  char* ws = (char*)d_ws;
  half_t* xh   = (half_t*)ws;  ws += (size_t)N_TOK * HID * 2;        // 25.2 MB
  half_t* w1ih = (half_t*)ws;  ws += (size_t)2048 * HID * 2;         // 3.1 MB
  half_t* w1oh = (half_t*)ws;  ws += (size_t)HID * 1024 * 2;         // 1.6 MB
  half_t* w2ih = (half_t*)ws;  ws += (size_t)4096 * HID * 2;         // 6.3 MB
  half_t* w2oh = (half_t*)ws;  ws += (size_t)HID * 2048 * 2;         // 3.1 MB
  half_t* Hbuf = (half_t*)ws;  ws += (size_t)N_TOK * 2048 * 2;       // 67.1 MB (reused per expert)
  float*  dwv  = (float*)ws;   ws += (size_t)N_TOK * 3 * 4;          // 0.2 MB
  float*  part = (float*)ws;   ws += (size_t)2048 * 2 * 4;           // partials

  // 1) fp32 -> fp16 staging
  cvt_f32_f16<<<2048, 256, 0, stream>>>(x,   xh,   N_TOK * HID);
  cvt_f32_f16<<<512,  256, 0, stream>>>(w1i, w1ih, 2048 * HID);
  cvt_f32_f16<<<512,  256, 0, stream>>>(w1o, w1oh, HID * 1024);
  cvt_f32_f16<<<512,  256, 0, stream>>>(w2i, w2ih, 4096 * HID);
  cvt_f32_f16<<<512,  256, 0, stream>>>(w2o, w2oh, HID * 2048);

  // 2) router + usage loss
  router_kernel<<<N_TOK / 8, 256, 0, stream>>>(x, pad, gate, logits, dwv, part);
  loss_reduce<<<1, 256, 0, stream>>>(part, N_TOK / 8, loss);

  // 3) out = dw0 * x
  init_out<<<8192, 256, 0, stream>>>(x, dwv, out, N_TOK * HID);

  // 4) expert 1 (h = 1024)
  gemm_swiglu<<<dim3(1024 / 64, N_TOK / 128), 256, 0, stream>>>(xh, w1ih, Hbuf, 1024);
  gemm_combine<<<dim3(HID / 64, N_TOK / 128), 256, 0, stream>>>(Hbuf, w1oh, dwv, out, 1024, 1);

  // 5) expert 2 (h = 2048), reuses Hbuf
  gemm_swiglu<<<dim3(2048 / 64, N_TOK / 128), 256, 0, stream>>>(xh, w2ih, Hbuf, 2048);
  gemm_combine<<<dim3(HID / 64, N_TOK / 128), 256, 0, stream>>>(Hbuf, w2oh, dwv, out, 2048, 2);
}
